// TriangleAttention_26242250179313
// MI455X (gfx1250) — compile-verified, matrix operand order checked by reading, not run on hardware
//
#include <hip/hip_runtime.h>
#include <math.h>

// ---------------------------------------------------------------------------
// TriangleAttention for MI455X (gfx1250), bf16 WMMA path.
// B=1, N=320, C=128, H=4, DC=32.
// ---------------------------------------------------------------------------

#define NRES 320
#define CCH  128
#define NH   4
#define DCH  32
#define TPR  20    // tiles per row: NRES/16

typedef __attribute__((ext_vector_type(16))) __bf16 v16bf;
typedef __attribute__((ext_vector_type(8)))  float  v8f;
typedef unsigned short bfraw;   // raw bf16 storage (avoid scalar __bf16 math)

union FragB { v16bf v; unsigned u[8]; };

__device__ __forceinline__ unsigned short f2bf(float f) {
    union { float f; unsigned u; } x; x.f = f;
    unsigned r = x.u + 0x7FFFu + ((x.u >> 16) & 1u);   // round-to-nearest-even
    return (unsigned short)(r >> 16);
}
__device__ __forceinline__ float bf2f(unsigned short h) {
    union { unsigned u; float f; } x; x.u = ((unsigned)h) << 16; return x.f;
}
__device__ __forceinline__ unsigned pack2(float a, float b) {
    return (unsigned)f2bf(a) | ((unsigned)f2bf(b) << 16);
}
__device__ __forceinline__ v8f wmma_bf16(v16bf a, v16bf b, v8f c) {
    return __builtin_amdgcn_wmma_f32_16x16x32_bf16(
        /*neg_a=*/false, a, /*neg_b=*/false, b,
        /*c_mod=*/(short)0, c, /*reuse_a=*/false, /*reuse_b=*/false);
}

// ---------------------------------------------------------------------------
// Kernel 0: pre-pack weights to bf16, TRANSPOSED to [col][k] so B-fragments
// are contiguous uint-pair loads (same pattern as A-fragments). wb padded to
// 16 columns with zeros -> no predication anywhere in the GEMM loops.
// Grid: (128, 6) x 128 threads.
// ---------------------------------------------------------------------------
__global__ __launch_bounds__(128) void pack_w_kernel(
    const float* __restrict__ wq, const float* __restrict__ wk,
    const float* __restrict__ wv, const float* __restrict__ wg,
    const float* __restrict__ wo, const float* __restrict__ wb,
    bfraw* __restrict__ qt, bfraw* __restrict__ kt, bfraw* __restrict__ vtw,
    bfraw* __restrict__ gt, bfraw* __restrict__ ot, bfraw* __restrict__ btw)
{
    const int col = blockIdx.x;       // 0..127
    const int mat = blockIdx.y;       // 0..5
    const int k   = threadIdx.x;      // 0..127
    if (mat < 5) {
        const float* W = (mat == 0) ? wq : (mat == 1) ? wk : (mat == 2) ? wv
                       : (mat == 3) ? wg : wo;
        bfraw* T = (mat == 0) ? qt : (mat == 1) ? kt : (mat == 2) ? vtw
                 : (mat == 3) ? gt : ot;
        T[(size_t)col * 128 + k] = f2bf(W[(size_t)k * 128 + col]);
    } else if (col < 16) {
        btw[(size_t)col * 128 + k] = (col < 4) ? f2bf(wb[(size_t)k * 4 + col])
                                               : (bfraw)0;
    }
}

// ---------------------------------------------------------------------------
// Kernel 1: LayerNorm + fused projections (q*scale, k, v^T, sigmoid(g), bterm)
// Grid: 6400 blocks x 128 threads (4 waves). Each block = 16 positions.
// A 16-position tile never crosses a row of the NxN grid (320 % 16 == 0), so
// (m, j0) are scalar per block -- no per-element division in the epilogue.
// ---------------------------------------------------------------------------
__global__ __launch_bounds__(128) void ln_proj_kernel(
    const float* __restrict__ x2d, const float* __restrict__ ln_g, const float* __restrict__ ln_b,
    const bfraw* __restrict__ wqt, const bfraw* __restrict__ wkt, const bfraw* __restrict__ wvt,
    const bfraw* __restrict__ wbt, const bfraw* __restrict__ wgt, const float* __restrict__ bg,
    bfraw* __restrict__ qb, bfraw* __restrict__ kb, bfraw* __restrict__ vt,
    bfraw* __restrict__ gb, float* __restrict__ bt)
{
    __shared__ unsigned xn_s[16][68];      // 16 rows x 128 bf16 (as uint pairs), padded
    __shared__ float red_s[16][8], red_q[16][8];
    __shared__ float mu_s[16], rs_s[16];

    const int tid   = threadIdx.x;
    const int ptile = blockIdx.x;
    const int r     = tid >> 3;            // row 0..15
    const int seg   = tid & 7;             // 16-channel segment
    const long p0   = (long)ptile * 16;
    const int  mrow = ptile / TPR;         // scalar: row of the NxN grid
    const int  j0   = (ptile % TPR) * 16;  // scalar: column base within the row

    // --- LayerNorm: partial sums per (row, segment) ---
    const float* xrow = x2d + (p0 + r) * CCH + seg * 16;
    float xv[16];
    float s = 0.f, sq = 0.f;
    #pragma unroll
    for (int i = 0; i < 16; i++) { float x = xrow[i]; xv[i] = x; s += x; sq += x * x; }
    red_s[r][seg] = s; red_q[r][seg] = sq;
    __syncthreads();
    if (tid < 16) {
        float ts = 0.f, tq = 0.f;
        #pragma unroll
        for (int k = 0; k < 8; k++) { ts += red_s[tid][k]; tq += red_q[tid][k]; }
        float mu  = ts * (1.f / 128.f);
        float var = tq * (1.f / 128.f) - mu * mu;
        mu_s[tid] = mu; rs_s[tid] = rsqrtf(var + 1e-5f);
    }
    __syncthreads();
    {
        float mu = mu_s[r], rs = rs_s[r];
        #pragma unroll
        for (int i = 0; i < 16; i += 2) {
            int c = seg * 16 + i;
            float a = (xv[i]     - mu) * rs * ln_g[c]     + ln_b[c];
            float b = (xv[i + 1] - mu) * rs * ln_g[c + 1] + ln_b[c + 1];
            xn_s[r][c >> 1] = pack2(a, b);
        }
    }
    __syncthreads();

    // Scalarize the wave id so tile dispatch becomes scalar branches.
    const int wid  = __builtin_amdgcn_readfirstlane(tid >> 5);   // wave 0..3
    const int lane = tid & 31;
    const int ln16 = lane & 15;
    const int klo2 = (lane < 16) ? 0 : 4;  // uint-pair offset of K half
    const int hi8  = (lane < 16) ? 0 : 8;

    // A fragments (xn, bf16 16x32 layout), shared across all column tiles
    FragB afr[4];
    #pragma unroll
    for (int kc = 0; kc < 4; kc++) {
        #pragma unroll
        for (int v = 0; v < 4; v++) {
            afr[kc].u[v]     = xn_s[ln16][kc * 16 + klo2 + v];
            afr[kc].u[v + 4] = xn_s[ln16][kc * 16 + 8 + klo2 + v];
        }
    }

    const float scale = 0.17677669529663687f;  // 1/sqrt(32)

    // 33 column tiles: q(0-7) k(8-15) v(16-23) g(24-31) bias(32)
    for (int t = wid; t < 33; t += 4) {
        const bfraw* Wt; int col0;
        if      (t < 8)  { Wt = wqt; col0 = t * 16; }
        else if (t < 16) { Wt = wkt; col0 = (t - 8) * 16; }
        else if (t < 24) { Wt = wvt; col0 = (t - 16) * 16; }
        else if (t < 32) { Wt = wgt; col0 = (t - 24) * 16; }
        else             { Wt = wbt; col0 = 0; }

        const int col = col0 + ln16;
        const unsigned* wrow = (const unsigned*)(Wt + (size_t)col * 128);

        // Prefetch all 4 B-fragments (8x global_load_b128 in one clause),
        // then run the 4-WMMA chain back-to-back.
        FragB bfr[4];
        #pragma unroll
        for (int kc = 0; kc < 4; kc++) {
            #pragma unroll
            for (int v = 0; v < 4; v++) {
                bfr[kc].u[v]     = wrow[kc * 16 + klo2 + v];
                bfr[kc].u[v + 4] = wrow[kc * 16 + 8 + klo2 + v];
            }
        }
        v8f acc = {};
        #pragma unroll
        for (int kc = 0; kc < 4; kc++) acc = wmma_bf16(afr[kc].v, bfr[kc].v, acc);

        if (t < 8) {
            #pragma unroll
            for (int r8 = 0; r8 < 8; r8++)
                qb[(p0 + r8 + hi8) * CCH + col] = f2bf(acc[r8] * scale);
        } else if (t < 16) {
            #pragma unroll
            for (int r8 = 0; r8 < 8; r8++)
                kb[(p0 + r8 + hi8) * CCH + col] = f2bf(acc[r8]);
        } else if (t < 24) {
            const int hh = col >> 5, c = col & 31;
            bfraw* vcol = vt + (((size_t)mrow * NH + hh) * DCH + c) * NRES + j0;
            #pragma unroll
            for (int r8 = 0; r8 < 8; r8++)
                vcol[r8 + hi8] = f2bf(acc[r8]);
        } else if (t < 32) {
            const float bgc = bg[col];
            #pragma unroll
            for (int r8 = 0; r8 < 8; r8++) {
                float gv = 1.f / (1.f + __expf(-(acc[r8] + bgc)));
                gb[(p0 + r8 + hi8) * CCH + col] = f2bf(gv);
            }
        } else if (col < 4) {
            // bt is [H][j][i] (transposed); here i = mrow, j = j0 + r8 + hi8
            float* bcol = bt + ((size_t)col * NRES + j0) * NRES + mrow;
            #pragma unroll
            for (int r8 = 0; r8 < 8; r8++)
                bcol[(size_t)(r8 + hi8) * NRES] = acc[r8];
        }
    }
}

// ---------------------------------------------------------------------------
// Kernel 2: attention. Grid: 320*4 blocks (m,h) x 64 threads (2 waves).
// Each wave owns a 16-row query tile; DC=32 -> one WMMA per QK^T tile.
// exp() written to LDS as packed bf16 pairs; 1/sum folded into epilogue.
// ---------------------------------------------------------------------------
__global__ __launch_bounds__(64) void attn_kernel(
    const bfraw* __restrict__ qb, const bfraw* __restrict__ kb, const bfraw* __restrict__ vt,
    const bfraw* __restrict__ gb, const float* __restrict__ bt, const float* __restrict__ mask,
    bfraw* __restrict__ op)
{
    __shared__ float    sm[2][16][NRES];       // 40 KB raw scores
    __shared__ unsigned pb[2][16][NRES / 2];   // 20 KB exp() as packed bf16 pairs
    __shared__ float    rinv_s[2][16];

    const int tid  = threadIdx.x;
    const int w    = __builtin_amdgcn_readfirstlane(tid >> 5);
    const int lane = tid & 31;
    const int ln16 = lane & 15;
    const int klo2 = (lane < 16) ? 0 : 4;
    const int hi8  = (lane < 16) ? 0 : 8;
    const int m    = blockIdx.x >> 2;
    const int h    = blockIdx.x & 3;

    for (int it = w; it < 20; it += 2) {
        // Q A-fragment (16 x 32, K = DC)
        FragB qf;
        {
            const unsigned* qrow = (const unsigned*)
                (qb + ((size_t)(m * NRES + it * 16 + ln16)) * CCH + h * DCH);
            #pragma unroll
            for (int v = 0; v < 4; v++) { qf.u[v] = qrow[klo2 + v]; qf.u[v + 4] = qrow[8 + klo2 + v]; }
        }

        // scores: S = Q K^T (+ pair bias + mask bias) into LDS
        for (int jt = 0; jt < 20; jt++) {
            FragB kf;
            const unsigned* krow = (const unsigned*)
                (kb + ((size_t)(m * NRES + jt * 16 + ln16)) * CCH + h * DCH);
            #pragma unroll
            for (int v = 0; v < 4; v++) { kf.u[v] = krow[klo2 + v]; kf.u[v + 4] = krow[8 + klo2 + v]; }

            const int   j  = jt * 16 + ln16;
            const float mb = 100000.f * (mask[m * NRES + j] - 1.f);
            // bt is [H][j][i] -> the 8 rows for this lane are contiguous in i
            const float4* b4 = (const float4*)
                (bt + ((size_t)h * NRES + j) * NRES + it * 16 + hi8);
            const float4 blo = b4[0], bhi = b4[1];

            v8f sacc = {};
            sacc = wmma_bf16(qf.v, kf.v, sacc);

            sm[w][0 + hi8][j] = sacc[0] + blo.x + mb;
            sm[w][1 + hi8][j] = sacc[1] + blo.y + mb;
            sm[w][2 + hi8][j] = sacc[2] + blo.z + mb;
            sm[w][3 + hi8][j] = sacc[3] + blo.w + mb;
            sm[w][4 + hi8][j] = sacc[4] + bhi.x + mb;
            sm[w][5 + hi8][j] = sacc[5] + bhi.y + mb;
            sm[w][6 + hi8][j] = sacc[6] + bhi.z + mb;
            sm[w][7 + hi8][j] = sacc[7] + bhi.w + mb;
        }
        __syncthreads();

        // softmax over j (one row per lane 0..15): exp -> packed bf16 pairs
        if (lane < 16) {
            float mx = -1e30f;
            for (int j = 0; j < NRES; j++) mx = fmaxf(mx, sm[w][lane][j]);
            float sum = 0.f;
            for (int j = 0; j < NRES; j += 2) {
                float e0 = __expf(sm[w][lane][j]     - mx);
                float e1 = __expf(sm[w][lane][j + 1] - mx);
                sum += e0 + e1;
                pb[w][lane][j >> 1] = pack2(e0, e1);
            }
            rinv_s[w][lane] = 1.f / sum;
        }
        __syncthreads();

        // PV: out[16 x 32] = E[16 x 320] @ V[320 x 32], K chunked by 32
        v8f acc0 = {}, acc1 = {};
        for (int jc = 0; jc < 10; jc++) {
            FragB pf, vf0, vf1;
            const unsigned* vrow0 = (const unsigned*)
                (vt + (((size_t)m * NH + h) * DCH + ln16)      * NRES + jc * 32);
            const unsigned* vrow1 = (const unsigned*)
                (vt + (((size_t)m * NH + h) * DCH + 16 + ln16) * NRES + jc * 32);
            #pragma unroll
            for (int v = 0; v < 4; v++) {
                vf0.u[v] = vrow0[klo2 + v]; vf0.u[v + 4] = vrow0[8 + klo2 + v];
                vf1.u[v] = vrow1[klo2 + v]; vf1.u[v + 4] = vrow1[8 + klo2 + v];
                pf.u[v]     = pb[w][ln16][jc * 16 + klo2 + v];
                pf.u[v + 4] = pb[w][ln16][jc * 16 + 8 + klo2 + v];
            }
            acc0 = wmma_bf16(pf.v, vf0.v, acc0);
            acc1 = wmma_bf16(pf.v, vf1.v, acc1);
        }

        // fold 1/sum, gate, store bf16 out_pre [p][h*32+c]
        #pragma unroll
        for (int r8 = 0; r8 < 8; r8++) {
            int i = it * 16 + r8 + hi8;
            size_t p   = (size_t)m * NRES + i;
            float rinv = rinv_s[w][r8 + hi8];
            float g0 = bf2f(gb[p * CCH + h * DCH + ln16]);
            float g1 = bf2f(gb[p * CCH + h * DCH + 16 + ln16]);
            op[p * CCH + h * DCH + ln16]      = f2bf(acc0[r8] * rinv * g0);
            op[p * CCH + h * DCH + 16 + ln16] = f2bf(acc1[r8] * rinv * g1);
        }
        __syncthreads();
    }
}

// ---------------------------------------------------------------------------
// Kernel 3: output projection [102400 x 128] @ wo[128 x 128] + bo -> f32 out
// Grid: 1600 blocks x 128 threads (4 waves, one 16-row tile each).
// ---------------------------------------------------------------------------
__global__ __launch_bounds__(128) void outproj_kernel(
    const bfraw* __restrict__ op, const bfraw* __restrict__ wot, const float* __restrict__ bo,
    float* __restrict__ out)
{
    const int tid  = threadIdx.x;
    const int w    = __builtin_amdgcn_readfirstlane(tid >> 5);
    const int lane = tid & 31;
    const int ln16 = lane & 15;
    const int klo2 = (lane < 16) ? 0 : 4;
    const int hi8  = (lane < 16) ? 0 : 8;
    const size_t ptile = (size_t)blockIdx.x * 4 + w;
    const size_t p0    = ptile * 16;

    FragB afr[4];
    const unsigned* arow = (const unsigned*)(op + (p0 + ln16) * CCH);
    #pragma unroll
    for (int kc = 0; kc < 4; kc++) {
        #pragma unroll
        for (int v = 0; v < 4; v++) {
            afr[kc].u[v]     = arow[kc * 16 + klo2 + v];
            afr[kc].u[v + 4] = arow[kc * 16 + 8 + klo2 + v];
        }
    }

    for (int ct = 0; ct < 8; ct++) {
        const int col = ct * 16 + ln16;
        const unsigned* wrow = (const unsigned*)(wot + (size_t)col * 128);
        FragB bfr[4];
        #pragma unroll
        for (int kc = 0; kc < 4; kc++) {
            #pragma unroll
            for (int v = 0; v < 4; v++) {
                bfr[kc].u[v]     = wrow[kc * 16 + klo2 + v];
                bfr[kc].u[v + 4] = wrow[kc * 16 + 8 + klo2 + v];
            }
        }
        v8f acc = {};
        #pragma unroll
        for (int kc = 0; kc < 4; kc++) acc = wmma_bf16(afr[kc].v, bfr[kc].v, acc);

        const float bias = bo[col];
        #pragma unroll
        for (int r8 = 0; r8 < 8; r8++) {
            out[(p0 + r8 + hi8) * CCH + col] = acc[r8] + bias;
        }
    }
}

// ---------------------------------------------------------------------------
extern "C" void kernel_launch(void* const* d_in, const int* in_sizes, int n_in,
                              void* d_out, int out_size, void* d_ws, size_t ws_size,
                              hipStream_t stream)
{
    const float* x2d  = (const float*)d_in[0];
    const float* mask = (const float*)d_in[1];
    const float* ln_g = (const float*)d_in[2];
    const float* ln_b = (const float*)d_in[3];
    const float* wq   = (const float*)d_in[4];
    const float* wk   = (const float*)d_in[5];
    const float* wv   = (const float*)d_in[6];
    const float* wb   = (const float*)d_in[7];
    const float* wg   = (const float*)d_in[8];
    const float* bg   = (const float*)d_in[9];
    const float* wo   = (const float*)d_in[10];
    const float* bo   = (const float*)d_in[11];

    char* ws = (char*)d_ws;
    const size_t npos  = (size_t)NRES * NRES;               // 102400
    const size_t sz_bf = npos * CCH * sizeof(bfraw);        // 26,214,400 B

    bfraw* qb = (bfraw*)(ws);
    bfraw* kb = (bfraw*)(ws + 1 * sz_bf);
    bfraw* vt = (bfraw*)(ws + 2 * sz_bf);
    bfraw* gb = (bfraw*)(ws + 3 * sz_bf);
    bfraw* op = (bfraw*)(ws + 4 * sz_bf);
    float* bt = (float*)(ws + 5 * sz_bf);                   // [H][N][N] f32 (j-major)

    char* wpk = ws + 5 * sz_bf + (size_t)NH * NRES * NRES * sizeof(float);
    bfraw* wqt = (bfraw*)(wpk);                             // [128][128] bf16 each
    bfraw* wkt = (bfraw*)(wpk + 1 * 32768);
    bfraw* wvt = (bfraw*)(wpk + 2 * 32768);
    bfraw* wgt = (bfraw*)(wpk + 3 * 32768);
    bfraw* wot = (bfraw*)(wpk + 4 * 32768);
    bfraw* wbt = (bfraw*)(wpk + 5 * 32768);                 // [16][128] bf16 (padded)

    pack_w_kernel<<<dim3(128, 6), 128, 0, stream>>>(
        wq, wk, wv, wg, wo, wb, wqt, wkt, wvt, wgt, wot, wbt);

    ln_proj_kernel<<<(int)(npos / 16), 128, 0, stream>>>(
        x2d, ln_g, ln_b, wqt, wkt, wvt, wbt, wgt, bg, qb, kb, vt, gb, bt);

    attn_kernel<<<NRES * NH, 64, 0, stream>>>(qb, kb, vt, gb, bt, mask, op);

    outproj_kernel<<<(int)(npos / 16 / 4), 128, 0, stream>>>(op, wot, bo, (float*)d_out);
}
